// DefenseTrajectoryTransformer_60258391163317
// MI455X (gfx1250) — matrix operand order repaired and checked by reading, not compiled, not run on hardware
//
#include <hip/hip_runtime.h>
#include <hip/hip_bf16.h>
#include <math.h>

typedef unsigned short u16;
typedef unsigned int   u32;
typedef __attribute__((ext_vector_type(16))) __bf16 v16bf;
typedef __attribute__((ext_vector_type(8)))  float  v8f;

union Frag16 { v16bf bf; uint4 q[2]; u16 h[16]; };
union AccF   { v8f v; float f[8]; };

__device__ __forceinline__ u16 f2bf(float f) {
  u32 u = __builtin_bit_cast(u32, f);
  u32 r = u + 0x7FFFu + ((u >> 16) & 1u);
  return (u16)(r >> 16);
}
__device__ __forceinline__ float bf2f(u16 h) {
  u32 u = ((u32)h) << 16;
  return __builtin_bit_cast(float, u);
}

// ---------------------------------------------------------------------------
// Weight packing: W (Nreal x Kreal row-major, y = x @ W^T) -> bf16 B-fragments
// layout: [nt][kt][lane(32)][j(16)],  n = nt*16 + lane%16,  k = kt*32 + (lane>=16?16:0) + j
// ---------------------------------------------------------------------------
__global__ void pack_w(const float* __restrict__ W, u16* __restrict__ out,
                       int Nreal, int Kreal, int Npad, int Kpad) {
  int KT = Kpad >> 5;
  size_t total = (size_t)Npad * Kpad;
  for (size_t idx = (size_t)blockIdx.x * 256 + threadIdx.x; idx < total;
       idx += (size_t)gridDim.x * 256) {
    int j  = (int)(idx & 15);
    int l  = (int)((idx >> 4) & 31);
    int kt = (int)((idx >> 9) % (size_t)KT);
    int nt = (int)(idx / (size_t)(512 * KT));
    int n = nt * 16 + (l & 15);
    int k = kt * 32 + ((l >= 16) ? 16 : 0) + j;
    float v = (n < Nreal && k < Kreal) ? W[(size_t)n * Kreal + k] : 0.f;
    out[idx] = f2bf(v);
  }
}

// A-fragment k map: k = kt*32 + j + (lane>=16 ? 16 : 0)
__device__ __forceinline__ int kmapA(int l, int j) {
  return j + ((l >= 16) ? 16 : 0);
}

// ---------------------------------------------------------------------------
// Generic encoder GEMM: out[M=32 rows/block][N] = A @ packedW^T (+epilogue)
// 256 thr = 8 waves; wave w: mtile = w/4, 4 ntiles (w%4)*4.. ; grid.y = N/256
// EPI: 0=bias  1=bias+posenc  2=bias+gelu  3=bias+residual+layernorm (N==256)
// ---------------------------------------------------------------------------
template <int EPI>
__global__ void __launch_bounds__(256) gemm_enc(
    const float* __restrict__ A, int lda, int Kreal, int Kpad,
    const u16* __restrict__ Bp, const float* __restrict__ bias,
    const float* __restrict__ res, const float* __restrict__ lnw,
    const float* __restrict__ lnb, float* __restrict__ out, int ldo) {
  extern __shared__ char smemraw[];
  u16* afrag = (u16*)smemraw;  // [2][KT][32][16]
  __shared__ float sumpart[8][16], sqpart[8][16], lnmean[32], lnrstd[32];

  const int KT = Kpad >> 5;
  const int tid = threadIdx.x;
  const size_t row0 = (size_t)blockIdx.x * 32;

  // cooperative A pack (fp32 -> bf16 A-fragments)
  int total = 2 * KT * 512;
  for (int idx = tid; idx < total; idx += 256) {
    int j = idx & 15, l = (idx >> 4) & 31;
    int kt = (idx >> 9) % KT, mt = idx / (512 * KT);
    int r = mt * 16 + (l & 15);
    int k = kt * 32 + kmapA(l, j);
    float v = (k < Kreal) ? A[(row0 + r) * (size_t)lda + k] : 0.f;
    afrag[idx] = f2bf(v);
  }
  __syncthreads();

  const int w = tid >> 5, lane = tid & 31;
  const int mt = w >> 2;
  const int ntbase = (w & 3) * 4;
  const int colchunk = blockIdx.y * 16;  // ntile offset

  AccF acc[4];
#pragma unroll
  for (int i = 0; i < 4; i++)
#pragma unroll
    for (int q = 0; q < 8; q++) acc[i].f[q] = 0.f;

  for (int kt = 0; kt < KT; ++kt) {
    Frag16 a;
    const u16* ap = afrag + ((size_t)(mt * KT + kt) * 32 + lane) * 16;
    a.q[0] = *(const uint4*)ap;
    a.q[1] = *(const uint4*)(ap + 8);
#pragma unroll
    for (int i = 0; i < 4; i++) {
      int ntg = colchunk + ntbase + i;
      const u16* bp = Bp + (((size_t)ntg * KT + kt) * 32 + lane) * 16;
      __builtin_prefetch(bp + 512, 0, 1);
      Frag16 b;
      b.q[0] = *(const uint4*)bp;
      b.q[1] = *(const uint4*)(bp + 8);
      acc[i].v = __builtin_amdgcn_wmma_f32_16x16x32_bf16(
          false, a.bf, false, b.bf, (short)0, acc[i].v, false, false);
    }
  }

  const int half8 = (lane >= 16) ? 8 : 0;
  if (EPI == 3) {
    float e[4][8];
#pragma unroll
    for (int i = 0; i < 4; i++) {
      int c = (colchunk + ntbase + i) * 16 + (lane & 15);
      float bb = bias[c];
#pragma unroll
      for (int v = 0; v < 8; v++) {
        size_t grow = row0 + mt * 16 + half8 + v;
        e[i][v] = acc[i].f[v] + bb + res[grow * (size_t)ldo + c];
      }
    }
#pragma unroll
    for (int v = 0; v < 8; v++) {
      float ps = 0.f, pq = 0.f;
#pragma unroll
      for (int i = 0; i < 4; i++) { ps += e[i][v]; pq += e[i][v] * e[i][v]; }
#pragma unroll
      for (int m = 1; m < 16; m <<= 1) {
        ps += __shfl_xor(ps, m, 32);
        pq += __shfl_xor(pq, m, 32);
      }
      if ((lane & 15) == 0) {
        int rl = v + half8;
        sumpart[w][rl] = ps;
        sqpart[w][rl] = pq;
      }
    }
    __syncthreads();
    if (tid < 32) {
      int mtt = tid >> 4, rl = tid & 15;
      float s = 0.f, s2 = 0.f;
      for (int q = 0; q < 4; q++) { s += sumpart[mtt * 4 + q][rl]; s2 += sqpart[mtt * 4 + q][rl]; }
      float mean = s * (1.f / 256.f);
      float var = s2 * (1.f / 256.f) - mean * mean;
      lnmean[tid] = mean;
      lnrstd[tid] = rsqrtf(var + 1e-5f);
    }
    __syncthreads();
#pragma unroll
    for (int i = 0; i < 4; i++) {
      int c = (colchunk + ntbase + i) * 16 + (lane & 15);
      float lw = lnw[c], lb = lnb[c];
#pragma unroll
      for (int v = 0; v < 8; v++) {
        int rloc = mt * 16 + half8 + v;
        size_t grow = row0 + rloc;
        out[grow * (size_t)ldo + c] = (e[i][v] - lnmean[rloc]) * lnrstd[rloc] * lw + lb;
      }
    }
  } else {
#pragma unroll
    for (int i = 0; i < 4; i++) {
      int c = (colchunk + ntbase + i) * 16 + (lane & 15);
      float bb = bias[c];
#pragma unroll
      for (int v = 0; v < 8; v++) {
        size_t grow = row0 + mt * 16 + half8 + v;
        float val = acc[i].f[v] + bb;
        if (EPI == 1) {  // sinusoidal positional encoding
          int trow = (int)(grow % 125);
          float ang = (float)trow * __expf((float)(c & ~1) * (-9.210340372f / 256.f));
          val += (c & 1) ? __cosf(ang) : __sinf(ang);
        } else if (EPI == 2) {  // exact gelu
          val = 0.5f * val * (1.f + erff(val * 0.70710678f));
        }
        out[grow * (size_t)ldo + c] = val;
      }
    }
  }
}

// ---------------------------------------------------------------------------
// Encoder attention: one block per (b,h). T=125 padded to 128.
// ---------------------------------------------------------------------------
__global__ void __launch_bounds__(256) attn_enc(const float* __restrict__ qkv,
                                                float* __restrict__ attnout) {
  __shared__ u16 qf[8 * 512], kf[8 * 512], vf[8 * 512], pf[8 * 4 * 512];
  int b = blockIdx.x >> 3, h = blockIdx.x & 7;
  int tid = threadIdx.x;
  const float* base = qkv + (size_t)b * 125 * 768 + h * 32;

  for (int idx = tid; idx < 4096; idx += 256) {  // Q A-fragments
    int j = idx & 15, l = (idx >> 4) & 31, mt = idx >> 9;
    int t = mt * 16 + (l & 15);
    int d = kmapA(l, j);
    qf[idx] = (t < 125) ? f2bf(base[(size_t)t * 768 + d]) : (u16)0;
  }
  for (int idx = tid; idx < 4096; idx += 256) {  // K B-fragments (scores)
    int j = idx & 15, l = (idx >> 4) & 31, nt = idx >> 9;
    int t = nt * 16 + (l & 15);
    int d = ((l >= 16) ? 16 : 0) + j;
    kf[idx] = (t < 125) ? f2bf(base[256 + (size_t)t * 768 + d]) : (u16)0;
  }
  for (int idx = tid; idx < 4096; idx += 256) {  // V B-fragments (PV)
    int j = idx & 15, l = (idx >> 4) & 31, nt = (idx >> 9) & 1, kt = idx >> 10;
    int t = kt * 32 + ((l >= 16) ? 16 : 0) + j;
    int d = nt * 16 + (l & 15);
    vf[idx] = (t < 125) ? f2bf(base[512 + (size_t)t * 768 + d]) : (u16)0;
  }
  __syncthreads();

  int w = tid >> 5, lane = tid & 31;
  AccF s[8];
  {
    Frag16 a;
    const u16* ap = qf + ((size_t)(w * 32 + lane)) * 16;
    a.q[0] = *(const uint4*)ap;
    a.q[1] = *(const uint4*)(ap + 8);
#pragma unroll
    for (int nt = 0; nt < 8; nt++) {
      AccF z;
#pragma unroll
      for (int q = 0; q < 8; q++) z.f[q] = 0.f;
      Frag16 bk;
      const u16* bp = kf + ((size_t)(nt * 32 + lane)) * 16;
      bk.q[0] = *(const uint4*)bp;
      bk.q[1] = *(const uint4*)(bp + 8);
      z.v = __builtin_amdgcn_wmma_f32_16x16x32_bf16(false, a.bf, false, bk.bf,
                                                    (short)0, z.v, false, false);
      s[nt] = z;
    }
  }
  // scale + mask + softmax (row lives in one half-wave)
  const float scale = 0.17677669529f;  // 1/sqrt(32)
  float rmax[8], rsum[8];
#pragma unroll
  for (int v = 0; v < 8; v++) rmax[v] = -1e30f;
#pragma unroll
  for (int nt = 0; nt < 8; nt++) {
    int col = nt * 16 + (lane & 15);
#pragma unroll
    for (int v = 0; v < 8; v++) {
      float x = (col < 125) ? s[nt].f[v] * scale : -1e30f;
      s[nt].f[v] = x;
      rmax[v] = fmaxf(rmax[v], x);
    }
  }
#pragma unroll
  for (int v = 0; v < 8; v++) {
#pragma unroll
    for (int m = 1; m < 16; m <<= 1) rmax[v] = fmaxf(rmax[v], __shfl_xor(rmax[v], m, 32));
    rsum[v] = 0.f;
  }
#pragma unroll
  for (int nt = 0; nt < 8; nt++)
#pragma unroll
    for (int v = 0; v < 8; v++) {
      float e = __expf(s[nt].f[v] - rmax[v]);
      s[nt].f[v] = e;
      rsum[v] += e;
    }
#pragma unroll
  for (int v = 0; v < 8; v++) {
#pragma unroll
    for (int m = 1; m < 16; m <<= 1) rsum[v] += __shfl_xor(rsum[v], m, 32);
    rsum[v] = 1.f / rsum[v];
  }
  // write probabilities as A-fragments (wave-local rows)
  int half8 = (lane >= 16) ? 8 : 0;
#pragma unroll
  for (int nt = 0; nt < 8; nt++) {
#pragma unroll
    for (int v = 0; v < 8; v++) {
      int r = v + half8;               // local row 0..15
      int c = nt * 16 + (lane & 15);   // key col 0..127
      int q = c & 31;
      int bsel = q >> 4;
      int j2 = q & 15;
      pf[(((size_t)w * 4 + (c >> 5)) * 32 + (r + 16 * bsel)) * 16 + j2] =
          f2bf(s[nt].f[v] * rsum[v]);
    }
  }
  __syncthreads();
  // PV: (128x128) @ (128x32)
#pragma unroll
  for (int nt2 = 0; nt2 < 2; nt2++) {
    AccF o;
#pragma unroll
    for (int q = 0; q < 8; q++) o.f[q] = 0.f;
#pragma unroll
    for (int kt = 0; kt < 4; kt++) {
      Frag16 pa, vb;
      const u16* ap = pf + ((size_t)(w * 4 + kt) * 32 + lane) * 16;
      pa.q[0] = *(const uint4*)ap;
      pa.q[1] = *(const uint4*)(ap + 8);
      const u16* bp = vf + ((size_t)(kt * 2 + nt2) * 32 + lane) * 16;
      vb.q[0] = *(const uint4*)bp;
      vb.q[1] = *(const uint4*)(bp + 8);
      o.v = __builtin_amdgcn_wmma_f32_16x16x32_bf16(false, pa.bf, false, vb.bf,
                                                    (short)0, o.v, false, false);
    }
#pragma unroll
    for (int v = 0; v < 8; v++) {
      int t = w * 16 + v + half8;
      if (t < 125) {
        int d = nt2 * 16 + (lane & 15);
        attnout[((size_t)b * 125 + t) * 256 + h * 32 + d] = o.f[v];
      }
    }
  }
}

// (B*T,256) fp32 -> bf16 [b][h][t][32]
__global__ void reorder_kv(const float* __restrict__ in, u16* __restrict__ out) {
  size_t idx = (size_t)blockIdx.x * 256 + threadIdx.x;
  if (idx < 8192000ull) {
    int d = (int)(idx & 31);
    size_t r = idx >> 5;
    int t = (int)(r % 125);
    size_t bh = r / 125;
    int h = (int)(bh & 7), b = (int)(bh >> 3);
    out[idx] = f2bf(in[((size_t)b * 125 + t) * 256 + h * 32 + d]);
  }
}

// ---------------------------------------------------------------------------
// Decoder: persistent kernel. 16 blocks x 256 thr; block owns 16 batch rows;
// loops 125 steps x 4 layers. All GEMMs via WMMA (M=16 tile).
// ---------------------------------------------------------------------------
struct DecParams {
  const u16 *w_din, *w_sav, *w_sao, *w_caq, *w_cao, *w_l1, *w_l2, *w_o1, *w_o2;
  const float *b_din, *sa_qkv_b, *sa_o_b, *ca_qkv_b, *ca_o_b, *l1_b, *l2_b;
  const float *ln1w, *ln1b, *ln2w, *ln2b, *ln3w, *ln3b, *out_b1, *out_b2;
  const u16 *Kc, *Vc;
  float* out;
};

__global__ void __launch_bounds__(256) dec_kernel(DecParams p) {
  extern __shared__ char sm[];
  float* y    = (float*)sm;                       // 16*256
  float* tmp  = y + 16 * 256;                     // 16*1024
  u16*   afrag = (u16*)(tmp + 16 * 1024);         // 32 ktiles * 512
  float* sc   = (float*)(afrag + 32 * 512);       // 128*128
  float* red  = sc + 128 * 128;                   // 512
  float* prev = red + 512;                        // 16*32
  float* lnr  = prev + 16 * 32;                   // 64

  const int tid = threadIdx.x, w = tid >> 5, lane = tid & 31;
  const int b0 = blockIdx.x * 16;

  auto packA = [&](const float* src, int ss, int Kreal, int KT) {
    int total = KT * 512;
    for (int idx = tid; idx < total; idx += 256) {
      int j = idx & 15, l = (idx >> 4) & 31, kt = idx >> 9;
      int r = l & 15;
      int k = kt * 32 + kmapA(l, j);
      afrag[idx] = (k < Kreal) ? f2bf(src[r * ss + k]) : (u16)0;
    }
  };

  auto gemm = [&](const u16* Bp, const float* bias, int biasN, float* out,
                  int NT, int KT, int act) {
    for (int nt = w; nt < NT; nt += 8) {
      AccF acc;
#pragma unroll
      for (int q = 0; q < 8; q++) acc.f[q] = 0.f;
      for (int kt = 0; kt < KT; ++kt) {
        Frag16 a, b;
        const u16* ap = afrag + ((size_t)(kt * 32 + lane)) * 16;
        a.q[0] = *(const uint4*)ap;
        a.q[1] = *(const uint4*)(ap + 8);
        const u16* bp = Bp + (((size_t)nt * KT + kt) * 32 + lane) * 16;
        __builtin_prefetch(bp + 512, 0, 1);
        b.q[0] = *(const uint4*)bp;
        b.q[1] = *(const uint4*)(bp + 8);
        acc.v = __builtin_amdgcn_wmma_f32_16x16x32_bf16(
            false, a.bf, false, b.bf, (short)0, acc.v, false, false);
      }
      int c = nt * 16 + (lane & 15);
      int rb = (lane >= 16) ? 8 : 0;
      float bb = (c < biasN) ? bias[c] : 0.f;
#pragma unroll
      for (int v = 0; v < 8; v++) {
        float val = acc.f[v] + bb;
        if (act == 1) val = 0.5f * val * (1.f + erff(val * 0.70710678f));
        else if (act == 2) val = fmaxf(val, 0.f);
        out[(rb + v) * 1024 + c] = val;
      }
    }
  };

  auto resLN = [&](const float* add, const float* lw, const float* lb) {
    __syncthreads();
    int r = tid >> 4, c0 = (tid & 15) * 16;
    float loc[16];
    float s = 0.f, s2 = 0.f;
#pragma unroll
    for (int i = 0; i < 16; i++) {
      float e = y[r * 256 + c0 + i] + add[r * 1024 + c0 + i];
      loc[i] = e; s += e; s2 += e * e;
    }
    sc[tid] = s; sc[256 + tid] = s2;
    __syncthreads();
    if (tid < 16) {
      float a = 0.f, b = 0.f;
      for (int i = 0; i < 16; i++) { a += sc[tid * 16 + i]; b += sc[256 + tid * 16 + i]; }
      float mean = a * (1.f / 256.f);
      float var = b * (1.f / 256.f) - mean * mean;
      lnr[tid] = mean; lnr[16 + tid] = rsqrtf(var + 1e-5f);
    }
    __syncthreads();
    float mean = lnr[r], rstd = lnr[16 + r];
#pragma unroll
    for (int i = 0; i < 16; i++) {
      int c = c0 + i;
      y[r * 256 + c] = (loc[i] - mean) * rstd * lw[c] + lb[c];
    }
    __syncthreads();
  };

  auto crossAttn = [&](int layer) {
    for (int idx = tid; idx < 4096; idx += 256)
      tmp[(idx >> 8) * 1024 + 256 + (idx & 255)] = 0.f;
    __syncthreads();
    int pair = tid >> 1, half = tid & 1;
    int r = pair >> 3, h = pair & 7;
    int bidx = b0 + r;
    const u16* Kb = p.Kc + ((((size_t)layer * 256 + bidx) * 8 + h) * 125) * 32;
    const u16* Vb = p.Vc + ((((size_t)layer * 256 + bidx) * 8 + h) * 125) * 32;
    float qv[32];
#pragma unroll
    for (int d = 0; d < 32; d++) qv[d] = tmp[r * 1024 + h * 32 + d];
    int start = half ? 63 : 0, end = half ? 125 : 63;
    float lmax = -1e30f;
    for (int t = start; t < end; t++) {
      const uint4* kp = (const uint4*)(Kb + (size_t)t * 32);
      float dot = 0.f;
#pragma unroll
      for (int i = 0; i < 4; i++) {
        uint4 kk = kp[i];
        u32 ww[4] = {kk.x, kk.y, kk.z, kk.w};
#pragma unroll
        for (int jj = 0; jj < 4; jj++) {
          dot += qv[i * 8 + 2 * jj] * bf2f((u16)(ww[jj] & 0xffff));
          dot += qv[i * 8 + 2 * jj + 1] * bf2f((u16)(ww[jj] >> 16));
        }
      }
      dot *= 0.17677669529f;
      sc[pair * 128 + t] = dot;
      lmax = fmaxf(lmax, dot);
    }
    red[pair * 2 + half] = lmax;
    __syncthreads();
    float m = fmaxf(red[pair * 2], red[pair * 2 + 1]);
    float lsum = 0.f;
    for (int t = start; t < end; t++) {
      float e = __expf(sc[pair * 128 + t] - m);
      sc[pair * 128 + t] = e;
      lsum += e;
    }
    red[256 + pair * 2 + half] = lsum;
    __syncthreads();
    float inv = 1.f / (red[256 + pair * 2] + red[256 + pair * 2 + 1]);
    float ctx[32];
#pragma unroll
    for (int d = 0; d < 32; d++) ctx[d] = 0.f;
    for (int t = start; t < end; t++) {
      float pw = sc[pair * 128 + t] * inv;
      const uint4* vp = (const uint4*)(Vb + (size_t)t * 32);
#pragma unroll
      for (int i = 0; i < 4; i++) {
        uint4 vv = vp[i];
        u32 ww[4] = {vv.x, vv.y, vv.z, vv.w};
#pragma unroll
        for (int jj = 0; jj < 4; jj++) {
          ctx[i * 8 + 2 * jj] += pw * bf2f((u16)(ww[jj] & 0xffff));
          ctx[i * 8 + 2 * jj + 1] += pw * bf2f((u16)(ww[jj] >> 16));
        }
      }
    }
#pragma unroll
    for (int d = 0; d < 32; d++)
      atomicAdd(&tmp[r * 1024 + 256 + h * 32 + d], ctx[d]);  // 2 adds: order-invariant
    __syncthreads();
  };

  // y init = positional encoding at position 0: sin(0)=0 / cos(0)=1
  for (int idx = tid; idx < 4096; idx += 256) y[idx] = (idx & 1) ? 1.f : 0.f;
  __syncthreads();

  for (int t = 0; t < 125; ++t) {
    for (int i = 0; i < 4; ++i) {
      // self-attention (len-1: softmax == 1 -> v-proj then o-proj)
      packA(y, 256, 256, 8); __syncthreads();
      gemm(p.w_sav + (size_t)i * 65536, p.sa_qkv_b + i * 768 + 512, 256, tmp, 16, 8, 0);
      __syncthreads();
      packA(tmp, 1024, 256, 8); __syncthreads();
      gemm(p.w_sao + (size_t)i * 65536, p.sa_o_b + i * 256, 256, tmp + 256, 16, 8, 0);
      __syncthreads();
      resLN(tmp + 256, p.ln1w + i * 256, p.ln1b + i * 256);
      // cross-attention
      packA(y, 256, 256, 8); __syncthreads();
      gemm(p.w_caq + (size_t)i * 65536, p.ca_qkv_b + i * 768, 256, tmp, 16, 8, 0);
      __syncthreads();
      crossAttn(i);
      packA(tmp + 256, 1024, 256, 8); __syncthreads();
      gemm(p.w_cao + (size_t)i * 65536, p.ca_o_b + i * 256, 256, tmp + 512, 16, 8, 0);
      __syncthreads();
      resLN(tmp + 512, p.ln2w + i * 256, p.ln2b + i * 256);
      // feed-forward
      packA(y, 256, 256, 8); __syncthreads();
      gemm(p.w_l1 + (size_t)i * 262144, p.l1_b + i * 1024, 1024, tmp, 64, 8, 1);
      __syncthreads();
      packA(tmp, 1024, 1024, 32); __syncthreads();
      gemm(p.w_l2 + (size_t)i * 262144, p.l2_b + i * 256, 256, tmp, 16, 32, 0);
      __syncthreads();
      resLN(tmp, p.ln3w + i * 256, p.ln3b + i * 256);
    }
    // output projection: relu(y@W1^T) @ W2^T
    packA(y, 256, 256, 8); __syncthreads();
    gemm(p.w_o1, p.out_b1, 64, tmp, 4, 8, 2); __syncthreads();
    packA(tmp, 1024, 64, 2); __syncthreads();
    gemm(p.w_o2, p.out_b2, 22, tmp + 64, 2, 2, 0); __syncthreads();
    for (int idx = tid; idx < 16 * 22; idx += 256) {
      int r = idx / 22, c = idx % 22;
      float v = tmp[r * 1024 + 64 + c];
      p.out[((size_t)(b0 + r) * 125 + t) * 22 + c] = v;
      prev[r * 32 + c] = v;
    }
    __syncthreads();
    if (t < 124) {  // din = prev @ Wdin^T + b + pe0
      packA(prev, 32, 22, 1); __syncthreads();
      gemm(p.w_din, p.b_din, 256, tmp, 16, 1, 0); __syncthreads();
      for (int idx = tid; idx < 4096; idx += 256) {
        int r = idx >> 8, c = idx & 255;
        y[r * 256 + c] = tmp[r * 1024 + c] + ((c & 1) ? 1.f : 0.f);
      }
      __syncthreads();
    }
  }
}

// ---------------------------------------------------------------------------
// Host side
// ---------------------------------------------------------------------------
extern "C" void kernel_launch(void* const* d_in, const int* in_sizes, int n_in,
                              void* d_out, int out_size, void* d_ws, size_t ws_size,
                              hipStream_t stream) {
  const float* cond      = (const float*)d_in[0];
  const float* enc_in_w  = (const float*)d_in[1];
  const float* enc_in_b  = (const float*)d_in[2];
  const float* dec_in_w  = (const float*)d_in[3];
  const float* dec_in_b  = (const float*)d_in[4];
  const float* e_qkv_w   = (const float*)d_in[5];
  const float* e_qkv_b   = (const float*)d_in[6];
  const float* e_o_w     = (const float*)d_in[7];
  const float* e_o_b     = (const float*)d_in[8];
  const float* e_l1_w    = (const float*)d_in[9];
  const float* e_l1_b    = (const float*)d_in[10];
  const float* e_l2_w    = (const float*)d_in[11];
  const float* e_l2_b    = (const float*)d_in[12];
  const float* e_ln1_w   = (const float*)d_in[13];
  const float* e_ln1_b   = (const float*)d_in[14];
  const float* e_ln2_w   = (const float*)d_in[15];
  const float* e_ln2_b   = (const float*)d_in[16];
  const float* sa_qkv_w  = (const float*)d_in[17];
  const float* sa_qkv_b  = (const float*)d_in[18];
  const float* sa_o_w    = (const float*)d_in[19];
  const float* sa_o_b    = (const float*)d_in[20];
  const float* ca_qkv_w  = (const float*)d_in[21];
  const float* ca_qkv_b  = (const float*)d_in[22];
  const float* ca_o_w    = (const float*)d_in[23];
  const float* ca_o_b    = (const float*)d_in[24];
  const float* dl1_w     = (const float*)d_in[25];
  const float* dl1_b     = (const float*)d_in[26];
  const float* dl2_w     = (const float*)d_in[27];
  const float* dl2_b     = (const float*)d_in[28];
  const float* dln1_w    = (const float*)d_in[29];
  const float* dln1_b    = (const float*)d_in[30];
  const float* dln2_w    = (const float*)d_in[31];
  const float* dln2_b    = (const float*)d_in[32];
  const float* dln3_w    = (const float*)d_in[33];
  const float* dln3_b    = (const float*)d_in[34];
  const float* out_w1    = (const float*)d_in[35];
  const float* out_b1    = (const float*)d_in[36];
  const float* out_w2    = (const float*)d_in[37];
  const float* out_b2    = (const float*)d_in[38];

  // packed-weight arena offsets (u16 elements)
  constexpr size_t PW_ENC_IN = 0;
  constexpr size_t PW_ENC_QKV = PW_ENC_IN + 256 * 160;
  constexpr size_t PW_ENC_O = PW_ENC_QKV + 4ull * 196608;
  constexpr size_t PW_ENC_L1 = PW_ENC_O + 4ull * 65536;
  constexpr size_t PW_ENC_L2 = PW_ENC_L1 + 4ull * 262144;
  constexpr size_t PW_DEC_IN = PW_ENC_L2 + 4ull * 262144;
  constexpr size_t PW_SAV = PW_DEC_IN + 8192;
  constexpr size_t PW_SAO = PW_SAV + 4ull * 65536;
  constexpr size_t PW_CAQ = PW_SAO + 4ull * 65536;
  constexpr size_t PW_CAK = PW_CAQ + 4ull * 65536;
  constexpr size_t PW_CAV = PW_CAK + 4ull * 65536;
  constexpr size_t PW_CAO = PW_CAV + 4ull * 65536;
  constexpr size_t PW_DL1 = PW_CAO + 4ull * 65536;
  constexpr size_t PW_DL2 = PW_DL1 + 4ull * 262144;
  constexpr size_t PW_O1 = PW_DL2 + 4ull * 262144;
  constexpr size_t PW_O2 = PW_O1 + 16384;
  constexpr size_t PW_END = PW_O2 + 2048;

  char* ws = (char*)d_ws;
  u16* pw = (u16*)ws;
  size_t fbase = ((PW_END * 2 + 255) / 256) * 256;
  float* X = (float*)(ws + fbase);        // 32000 x 256 (encoder state / memory)
  float* SCR = X + 8192000ull;            // 32000 x 1024 scratch
  float* QKV = SCR;                       // 32000 x 768
  float* ATT = SCR + 24576000ull;         // 32000 x 256
  float* H = SCR;                         // 32000 x 1024 (FF hidden, reuses SCR)
  u16* KC = (u16*)(SCR + 32768000ull);    // L x B x H x 125 x 32 bf16
  u16* VC = KC + 32768000ull;

  auto pack = [&](const float* W, size_t off, int Nr, int Kr, int Np, int Kp) {
    size_t total = (size_t)Np * Kp;
    int blocks = (int)((total + 255) / 256);
    if (blocks > 4096) blocks = 4096;
    pack_w<<<blocks, 256, 0, stream>>>(W, pw + off, Nr, Kr, Np, Kp);
  };

  pack(enc_in_w, PW_ENC_IN, 256, 158, 256, 160);
  pack(dec_in_w, PW_DEC_IN, 256, 22, 256, 32);
  pack(out_w1, PW_O1, 64, 256, 64, 256);
  pack(out_w2, PW_O2, 22, 64, 32, 64);
  for (int i = 0; i < 4; i++) {
    pack(e_qkv_w + (size_t)i * 196608, PW_ENC_QKV + (size_t)i * 196608, 768, 256, 768, 256);
    pack(e_o_w + (size_t)i * 65536, PW_ENC_O + (size_t)i * 65536, 256, 256, 256, 256);
    pack(e_l1_w + (size_t)i * 262144, PW_ENC_L1 + (size_t)i * 262144, 1024, 256, 1024, 256);
    pack(e_l2_w + (size_t)i * 262144, PW_ENC_L2 + (size_t)i * 262144, 256, 1024, 256, 1024);
    pack(sa_qkv_w + (size_t)i * 196608 + 512 * 256, PW_SAV + (size_t)i * 65536, 256, 256, 256, 256);
    pack(sa_o_w + (size_t)i * 65536, PW_SAO + (size_t)i * 65536, 256, 256, 256, 256);
    pack(ca_qkv_w + (size_t)i * 196608, PW_CAQ + (size_t)i * 65536, 256, 256, 256, 256);
    pack(ca_qkv_w + (size_t)i * 196608 + 256 * 256, PW_CAK + (size_t)i * 65536, 256, 256, 256, 256);
    pack(ca_qkv_w + (size_t)i * 196608 + 512 * 256, PW_CAV + (size_t)i * 65536, 256, 256, 256, 256);
    pack(ca_o_w + (size_t)i * 65536, PW_CAO + (size_t)i * 65536, 256, 256, 256, 256);
    pack(dl1_w + (size_t)i * 262144, PW_DL1 + (size_t)i * 262144, 1024, 256, 1024, 256);
    pack(dl2_w + (size_t)i * 262144, PW_DL2 + (size_t)i * 262144, 256, 1024, 256, 1024);
  }

  // ---- encoder ----
  gemm_enc<1><<<dim3(1000, 1), 256, 160 * 64, stream>>>(
      cond, 158, 158, 160, pw + PW_ENC_IN, enc_in_b, nullptr, nullptr, nullptr, X, 256);
  for (int i = 0; i < 4; i++) {
    gemm_enc<0><<<dim3(1000, 3), 256, 256 * 64, stream>>>(
        X, 256, 256, 256, pw + PW_ENC_QKV + (size_t)i * 196608, e_qkv_b + i * 768,
        nullptr, nullptr, nullptr, QKV, 768);
    attn_enc<<<2048, 256, 0, stream>>>(QKV, ATT);
    gemm_enc<3><<<dim3(1000, 1), 256, 256 * 64, stream>>>(
        ATT, 256, 256, 256, pw + PW_ENC_O + (size_t)i * 65536, e_o_b + i * 256,
        X, e_ln1_w + i * 256, e_ln1_b + i * 256, X, 256);
    gemm_enc<2><<<dim3(1000, 4), 256, 256 * 64, stream>>>(
        X, 256, 256, 256, pw + PW_ENC_L1 + (size_t)i * 262144, e_l1_b + i * 1024,
        nullptr, nullptr, nullptr, H, 1024);
    gemm_enc<3><<<dim3(1000, 1), 256, 1024 * 64, stream>>>(
        H, 1024, 1024, 1024, pw + PW_ENC_L2 + (size_t)i * 262144, e_l2_b + i * 256,
        X, e_ln2_w + i * 256, e_ln2_b + i * 256, X, 256);
  }
  // ---- precompute cross-attn K/V in bf16 (L2-resident) ----
  for (int i = 0; i < 4; i++) {
    gemm_enc<0><<<dim3(1000, 1), 256, 256 * 64, stream>>>(
        X, 256, 256, 256, pw + PW_CAK + (size_t)i * 65536, ca_qkv_b + i * 768 + 256,
        nullptr, nullptr, nullptr, SCR, 256);
    reorder_kv<<<32000, 256, 0, stream>>>(SCR, KC + (size_t)i * 8192000);
    gemm_enc<0><<<dim3(1000, 1), 256, 256 * 64, stream>>>(
        X, 256, 256, 256, pw + PW_CAV + (size_t)i * 65536, ca_qkv_b + i * 768 + 512,
        nullptr, nullptr, nullptr, SCR, 256);
    reorder_kv<<<32000, 256, 0, stream>>>(SCR, VC + (size_t)i * 8192000);
  }
  // ---- decoder (persistent, 125 recurrent steps) ----
  DecParams P;
  P.w_din = pw + PW_DEC_IN; P.w_sav = pw + PW_SAV; P.w_sao = pw + PW_SAO;
  P.w_caq = pw + PW_CAQ; P.w_cao = pw + PW_CAO; P.w_l1 = pw + PW_DL1;
  P.w_l2 = pw + PW_DL2; P.w_o1 = pw + PW_O1; P.w_o2 = pw + PW_O2;
  P.b_din = dec_in_b; P.sa_qkv_b = sa_qkv_b; P.sa_o_b = sa_o_b;
  P.ca_qkv_b = ca_qkv_b; P.ca_o_b = ca_o_b; P.l1_b = dl1_b; P.l2_b = dl2_b;
  P.ln1w = dln1_w; P.ln1b = dln1_b; P.ln2w = dln2_w; P.ln2b = dln2_b;
  P.ln3w = dln3_w; P.ln3b = dln3_b; P.out_b1 = out_b1; P.out_b2 = out_b2;
  P.Kc = KC; P.Vc = VC; P.out = (float*)d_out;
  size_t dec_sh = 16 * 256 * 4 + 16 * 1024 * 4 + 32 * 512 * 2 + 128 * 128 * 4 +
                  512 * 4 + 16 * 32 * 4 + 64 * 4;
  dec_kernel<<<16, 256, dec_sh, stream>>>(P);
  (void)in_sizes; (void)n_in; (void)out_size; (void)ws_size;
}
